// TopKTranscoder_31293131718912
// MI455X (gfx1250) — compile-verified, matrix operand order checked by reading, not run on hardware
//
#include <hip/hip_runtime.h>
#include <hip/hip_bf16.h>

// ---------------------------------------------------------------------------
// Problem constants (match the reference)
// ---------------------------------------------------------------------------
#define B_SZ   8192
#define D_IN   1024
#define D_OUT  1024
#define H_SZ   16384
#define K_TOP  32

typedef __attribute__((ext_vector_type(16))) __bf16 v16bf;
typedef __attribute__((ext_vector_type(8)))  float  v8f;

union Frag16 {
    v16bf v;
    uint4 u[2];
};

// Async global->LDS staging (ASYNCcnt path), guarded so we always compile.
#if defined(__AMDGCN__) && __has_builtin(__builtin_amdgcn_global_load_async_to_lds_b128) && __has_builtin(__builtin_amdgcn_s_wait_asynccnt)
#define USE_ASYNC_COPY 1
#else
#define USE_ASYNC_COPY 0
#endif

#if USE_ASYNC_COPY
// Parameter types per hipcc diagnostic: pointer to `int __vector(4)` in the
// global (addrspace(1), printed "__device__") and LDS (addrspace(3)) spaces.
typedef int v4i_async __attribute__((vector_size(16)));
typedef __attribute__((address_space(1))) v4i_async* g_v4i_ptr;
typedef __attribute__((address_space(3))) v4i_async* l_v4i_ptr;

__device__ __forceinline__ void async_copy16(const void* gsrc, void* ldst) {
    __builtin_amdgcn_global_load_async_to_lds_b128(
        (g_v4i_ptr)(void*)gsrc,
        (l_v4i_ptr)ldst,
        0, 0);
}
#endif

// ---------------------------------------------------------------------------
// Kernel 1: f32 -> bf16 conversion (RNE via compiler cast)
// ---------------------------------------------------------------------------
__global__ __launch_bounds__(256) void cvt_bf16_kernel(const float* __restrict__ src,
                                                       __bf16* __restrict__ dst,
                                                       int n) {
    int i = blockIdx.x * 256 + threadIdx.x;
    if (i < n) dst[i] = (__bf16)src[i];
}

// ---------------------------------------------------------------------------
// Kernel 2: tiled transpose W_dec [D_OUT, H] -> W_decT [H, D_OUT]
// ---------------------------------------------------------------------------
__global__ __launch_bounds__(256) void transpose_wdec_kernel(const float* __restrict__ src,
                                                             float* __restrict__ dst) {
    __shared__ float tile[32][33];
    const int hBase = blockIdx.x * 32;   // H dimension
    const int dBase = blockIdx.y * 32;   // D_OUT dimension
    const int tx = threadIdx.x;          // 0..31
    const int ty = threadIdx.y;          // 0..7
#pragma unroll
    for (int i = 0; i < 32; i += 8)
        tile[ty + i][tx] = src[(size_t)(dBase + ty + i) * H_SZ + hBase + tx];
    __syncthreads();
#pragma unroll
    for (int i = 0; i < 32; i += 8)
        dst[(size_t)(hBase + ty + i) * D_OUT + dBase + tx] = tile[tx][ty + i];
}

// ---------------------------------------------------------------------------
// Kernel 3: encoder GEMM  pre[B,H] = Xbf[B,K] * Wbf[H,K]^T + b_enc
// 128x128 block tile, 8 waves (2x4), wave tile 64x32, K-step 32 (bf16 WMMA).
// ---------------------------------------------------------------------------
#define BM 128
#define BN 128
#define BK 32
#define LDS_STRIDE 40   // (BK + 8) bf16 -> 80 bytes, multiple of 16B

__global__ __launch_bounds__(256) void encoder_gemm_kernel(const __bf16* __restrict__ aBf,
                                                           const __bf16* __restrict__ wBf,
                                                           const float* __restrict__ bEnc,
                                                           float* __restrict__ pre) {
    __shared__ __bf16 As[BM * LDS_STRIDE];
    __shared__ __bf16 Bs[BN * LDS_STRIDE];

    const int t     = threadIdx.x;
    const int lane  = t & 31;
    const int wid   = t >> 5;        // 0..7
    const int waveM = wid >> 2;      // 0..1  (64-row slabs)
    const int waveN = wid & 3;       // 0..3  (32-col slabs)

    const int mBase = blockIdx.y * BM;
    const int nBase = blockIdx.x * BN;

    const int lrow = lane & 15;
    const int kb   = (lane < 16) ? 0 : 8;   // bf16 A/B fragment K-chunk base

    const v8f vzero = {0.f, 0.f, 0.f, 0.f, 0.f, 0.f, 0.f, 0.f};
    v8f acc[4][2];
#pragma unroll
    for (int mi = 0; mi < 4; ++mi)
#pragma unroll
        for (int ni = 0; ni < 2; ++ni) acc[mi][ni] = vzero;

    const int ldRow = t >> 2;          // 0..63
    const int ldCol = (t & 3) * 8;     // 0,8,16,24 (bf16 elements)

    for (int k0 = 0; k0 < D_IN; k0 += BK) {
        // stage A (128x32) and B (128x32) tiles into LDS, 16B per access
#if USE_ASYNC_COPY
#pragma unroll
        for (int it = 0; it < 2; ++it) {
            const int r = ldRow + it * 64;
            async_copy16(aBf + (size_t)(mBase + r) * D_IN + k0 + ldCol,
                         &As[r * LDS_STRIDE + ldCol]);
            async_copy16(wBf + (size_t)(nBase + r) * D_IN + k0 + ldCol,
                         &Bs[r * LDS_STRIDE + ldCol]);
            if (k0 + BK < D_IN) {
                __builtin_prefetch(aBf + (size_t)(mBase + r) * D_IN + k0 + BK + ldCol, 0, 1);
                __builtin_prefetch(wBf + (size_t)(nBase + r) * D_IN + k0 + BK + ldCol, 0, 1);
            }
        }
        __builtin_amdgcn_s_wait_asynccnt(0);
        __syncthreads();
#else
#pragma unroll
        for (int it = 0; it < 2; ++it) {
            const int r = ldRow + it * 64;
            const uint4* pa = reinterpret_cast<const uint4*>(
                aBf + (size_t)(mBase + r) * D_IN + k0 + ldCol);
            *reinterpret_cast<uint4*>(&As[r * LDS_STRIDE + ldCol]) = *pa;
            const uint4* pb = reinterpret_cast<const uint4*>(
                wBf + (size_t)(nBase + r) * D_IN + k0 + ldCol);
            *reinterpret_cast<uint4*>(&Bs[r * LDS_STRIDE + ldCol]) = *pb;
            // prefetch next K tile
            if (k0 + BK < D_IN) {
                __builtin_prefetch(aBf + (size_t)(mBase + r) * D_IN + k0 + BK + ldCol, 0, 1);
                __builtin_prefetch(wBf + (size_t)(nBase + r) * D_IN + k0 + BK + ldCol, 0, 1);
            }
        }
        __syncthreads();
#endif

        Frag16 aF[4], bF[2];
#pragma unroll
        for (int mi = 0; mi < 4; ++mi) {
            const int rr = waveM * 64 + mi * 16 + lrow;
            aF[mi].u[0] = *reinterpret_cast<const uint4*>(&As[rr * LDS_STRIDE + kb]);
            aF[mi].u[1] = *reinterpret_cast<const uint4*>(&As[rr * LDS_STRIDE + kb + 16]);
        }
#pragma unroll
        for (int ni = 0; ni < 2; ++ni) {
            const int rr = waveN * 32 + ni * 16 + lrow;
            bF[ni].u[0] = *reinterpret_cast<const uint4*>(&Bs[rr * LDS_STRIDE + kb]);
            bF[ni].u[1] = *reinterpret_cast<const uint4*>(&Bs[rr * LDS_STRIDE + kb + 16]);
        }

#pragma unroll
        for (int mi = 0; mi < 4; ++mi)
#pragma unroll
            for (int ni = 0; ni < 2; ++ni)
                acc[mi][ni] = __builtin_amdgcn_wmma_f32_16x16x32_bf16(
                    false, aF[mi].v, false, bF[ni].v,
                    (short)0, acc[mi][ni], false, false);

        __syncthreads();
    }

    // write pre-activations (+bias) to global
#pragma unroll
    for (int mi = 0; mi < 4; ++mi) {
#pragma unroll
        for (int ni = 0; ni < 2; ++ni) {
            const int col  = nBase + waveN * 32 + ni * 16 + (lane & 15);
            const float bb = bEnc[col];
#pragma unroll
            for (int r = 0; r < 8; ++r) {
                const int rowg = mBase + waveM * 64 + mi * 16 + ((lane < 16) ? r : r + 8);
                pre[(size_t)rowg * H_SZ + col] = acc[mi][ni][r] + bb;
            }
        }
    }
}

// ---------------------------------------------------------------------------
// Kernel 4: exact per-row top-K via 4-pass radix select; rewrites hidden in
// place (relu of selected, 0 elsewhere), emits compact (idx,val) list and
// per-row l0 count. Fully deterministic (prefix-scan tie breaking).
// ---------------------------------------------------------------------------
__global__ __launch_bounds__(256) void topk_kernel(float* __restrict__ hid,
                                                   int* __restrict__ selIdx,
                                                   float* __restrict__ selVal,
                                                   int* __restrict__ rowCnt) {
    const int row = blockIdx.x;
    const int t   = threadIdx.x;
    float* rp = hid + (size_t)row * H_SZ;

    // monotone key transform: larger key <=> larger float
    unsigned key[64];
#pragma unroll 4
    for (int i = 0; i < 64; ++i) {
        const int col = t + (i << 8);
        unsigned s = __float_as_uint(rp[col]);
        key[i] = (s & 0x80000000u) ? ~s : (s | 0x80000000u);
    }

    __shared__ int hist[256];
    __shared__ unsigned prefixSh;
    __shared__ int kRemSh;
    if (t == 0) { prefixSh = 0u; kRemSh = K_TOP; }
    __syncthreads();

    unsigned pmask = 0u;
    for (int pass = 0; pass < 4; ++pass) {
        const int shift = 24 - 8 * pass;
        hist[t] = 0;
        __syncthreads();
        const unsigned prefix = prefixSh;
        for (int i = 0; i < 64; ++i)
            if ((key[i] & pmask) == prefix)
                atomicAdd(&hist[(key[i] >> shift) & 255], 1);
        __syncthreads();
        if (t == 0) {
            int cnt = 0, chosen = 0;
            const int kRem = kRemSh;
            for (int b = 255; b >= 0; --b) {
                const int c = hist[b];
                if (cnt + c >= kRem) { chosen = b; break; }
                cnt += c;
            }
            kRemSh  = kRem - cnt;
            prefixSh |= ((unsigned)chosen) << shift;
        }
        pmask |= 0xFFu << shift;
        __syncthreads();
    }

    const unsigned uthr = prefixSh;   // 32nd largest key
    const int Tties     = kRemSh;     // how many ties at threshold to keep

    // deterministic ranks for strictly-greater and tie elements
    int gc = 0, tc = 0;
    for (int i = 0; i < 64; ++i) {
        if (key[i] > uthr) ++gc;
        else if (key[i] == uthr) ++tc;
    }
    __shared__ int sG[256], sT[256];
    __shared__ int GtotSh;
    __shared__ int cntSh;
    sG[t] = gc; sT[t] = tc;
    if (t == 0) cntSh = 0;
    __syncthreads();
    if (t == 0) {
        int ag = 0, at = 0;
        for (int i = 0; i < 256; ++i) {
            const int g = sG[i]; sG[i] = ag; ag += g;
            const int x = sT[i]; sT[i] = at; at += x;
        }
        GtotSh = ag;
    }
    __syncthreads();

    int gpos = sG[t];
    int tpos = sT[t];
    const int Gtot = GtotSh;
    int pcnt = 0;

    for (int i = 0; i < 64; ++i) {
        const int col = t + (i << 8);
        const unsigned k = key[i];
        int slot = -1;
        if (k > uthr) {
            slot = gpos++;
        } else if (k == uthr) {
            if (tpos < Tties) slot = Gtot + tpos;
            ++tpos;
        }
        float outv = 0.f;
        if (slot >= 0) {
            const unsigned s = (k & 0x80000000u) ? (k ^ 0x80000000u) : ~k;
            const float v = __uint_as_float(s);
            outv = v > 0.f ? v : 0.f;
            selIdx[row * K_TOP + slot] = col;
            selVal[row * K_TOP + slot] = outv;
            if (outv > 0.f) ++pcnt;
        }
        rp[col] = outv;
    }
    atomicAdd(&cntSh, pcnt);   // integer: deterministic
    __syncthreads();
    if (t == 0) rowCnt[row] = cntSh;
}

// ---------------------------------------------------------------------------
// Kernel 5: sparse decode + MSE partial: pred[b,:] = b_dec + sum_k v_k*WdT[i_k]
// ---------------------------------------------------------------------------
__global__ __launch_bounds__(256) void decode_kernel(const float* __restrict__ wdT,
                                                     const float* __restrict__ bDec,
                                                     const int* __restrict__ selIdx,
                                                     const float* __restrict__ selVal,
                                                     const float* __restrict__ y,
                                                     float* __restrict__ pred,
                                                     float* __restrict__ partial) {
    const int row = blockIdx.x;
    const int t   = threadIdx.x;
    __shared__ int   sIdx[K_TOP];
    __shared__ float sVal[K_TOP];
    if (t < K_TOP) {
        sIdx[t] = selIdx[row * K_TOP + t];
        sVal[t] = selVal[row * K_TOP + t];
    }
    __syncthreads();

    float acc[4];
#pragma unroll
    for (int j = 0; j < 4; ++j) acc[j] = bDec[t + j * 256];

    for (int k = 0; k < K_TOP; ++k) {
        const float v = sVal[k];
        const float* wcol = wdT + (size_t)sIdx[k] * D_OUT;
#pragma unroll
        for (int j = 0; j < 4; ++j) acc[j] += v * wcol[t + j * 256];
    }

    float ss = 0.f;
#pragma unroll
    for (int j = 0; j < 4; ++j) {
        const int d = t + j * 256;
        const float p = acc[j];
        pred[(size_t)row * D_OUT + d] = p;
        const float df = p - y[(size_t)row * D_OUT + d];
        ss += df * df;
    }

    __shared__ float red[256];
    red[t] = ss;
    __syncthreads();
    for (int off = 128; off > 0; off >>= 1) {
        if (t < off) red[t] += red[t + off];
        __syncthreads();
    }
    if (t == 0) partial[row] = red[0];
}

// ---------------------------------------------------------------------------
// Kernel 6: final deterministic reduction -> 4 scalar outputs
// ---------------------------------------------------------------------------
__global__ __launch_bounds__(256) void finalize_kernel(const float* __restrict__ partial,
                                                       const int* __restrict__ rowCnt,
                                                       float* __restrict__ sc) {
    __shared__ double sd[256];
    __shared__ long long si[256];
    const int t = threadIdx.x;
    double s = 0.0;
    long long c = 0;
    for (int i = t; i < B_SZ; i += 256) { s += (double)partial[i]; c += rowCnt[i]; }
    sd[t] = s; si[t] = c;
    __syncthreads();
    for (int off = 128; off > 0; off >>= 1) {
        if (t < off) { sd[t] += sd[t + off]; si[t] += si[t + off]; }
        __syncthreads();
    }
    if (t == 0) {
        const float recon = (float)(sd[0] / ((double)B_SZ * (double)D_OUT));
        sc[0] = recon;                                  // loss
        sc[1] = recon;                                  // reconstruction_loss
        sc[2] = 0.f;                                    // sparsity_loss
        sc[3] = (float)((double)si[0] / (double)B_SZ);  // l0
    }
}

// ---------------------------------------------------------------------------
// Host-side orchestration
// ---------------------------------------------------------------------------
extern "C" void kernel_launch(void* const* d_in, const int* in_sizes, int n_in,
                              void* d_out, int out_size, void* d_ws, size_t ws_size,
                              hipStream_t stream) {
    const float* mlp_input  = (const float*)d_in[0];   // [B, D_IN]
    const float* mlp_output = (const float*)d_in[1];   // [B, D_OUT]
    const float* W_enc      = (const float*)d_in[2];   // [H, D_IN]
    const float* b_enc      = (const float*)d_in[3];   // [H]
    const float* W_dec      = (const float*)d_in[4];   // [D_OUT, H]
    const float* b_dec      = (const float*)d_in[5];   // [D_OUT]

    float* out_pred   = (float*)d_out;                                  // [B, D_OUT]
    float* out_hidden = (float*)d_out + (size_t)B_SZ * D_OUT;           // [B, H]
    float* out_scal   = (float*)d_out + (size_t)B_SZ * D_OUT + (size_t)B_SZ * H_SZ;

    // workspace layout (256B-aligned slabs)
    char* w = (char*)d_ws;
    __bf16* aBf    = (__bf16*)(w);                                      // 16 MB
    __bf16* wBf    = (__bf16*)(w + (size_t)16 * 1024 * 1024);           // 32 MB
    float*  wdT    = (float*) (w + (size_t)48 * 1024 * 1024);           // 64 MB
    int*    selIdx = (int*)   (w + (size_t)112 * 1024 * 1024);          //  1 MB
    float*  selVal = (float*) (w + (size_t)113 * 1024 * 1024);          //  1 MB
    float*  part   = (float*) (w + (size_t)114 * 1024 * 1024);          // 32 KB
    int*    rowCnt = (int*)   (w + (size_t)114 * 1024 * 1024 + 65536);  // 32 KB

    // 1) f32 -> bf16 for GEMM operands
    {
        const int nA = B_SZ * D_IN;
        cvt_bf16_kernel<<<(nA + 255) / 256, 256, 0, stream>>>(mlp_input, aBf, nA);
        const int nW = H_SZ * D_IN;
        cvt_bf16_kernel<<<(nW + 255) / 256, 256, 0, stream>>>(W_enc, wBf, nW);
    }

    // 2) transpose decoder weights for coalesced sparse gathers
    {
        dim3 grid(H_SZ / 32, D_OUT / 32);
        dim3 block(32, 8);
        transpose_wdec_kernel<<<grid, block, 0, stream>>>(W_dec, wdT);
    }

    // 3) encoder GEMM (WMMA bf16 -> f32), writes pre-activations into hidden slab
    {
        dim3 grid(H_SZ / BN, B_SZ / BM);
        encoder_gemm_kernel<<<grid, 256, 0, stream>>>(aBf, wBf, b_enc, out_hidden);
    }

    // 4) exact top-32 per row (radix select), rewrite hidden in place
    topk_kernel<<<B_SZ, 256, 0, stream>>>(out_hidden, selIdx, selVal, rowCnt);

    // 5) sparse decode + MSE partials
    decode_kernel<<<B_SZ, 256, 0, stream>>>(wdT, b_dec, selIdx, selVal,
                                            mlp_output, out_pred, part);

    // 6) scalars
    finalize_kernel<<<1, 256, 0, stream>>>(part, rowCnt, out_scal);

    (void)in_sizes; (void)n_in; (void)out_size; (void)ws_size;
}
